// SABlock_22067541967572
// MI455X (gfx1250) — matrix operand order repaired and checked by reading, not compiled, not run on hardware
//
#include <hip/hip_runtime.h>
#include <hip/hip_bf16.h>

// ---------------- problem constants ----------------
#define DIM_C  768
#define NTOK   1568           // 8*14*14
#define BATCH  4
#define MTOK   (BATCH * NTOK) // 6272
#define H3     2304           // 3*DIM_C
#define HID    3072
#define NHEAD  12
#define DHEAD  64
#define TOTAL  (BATCH * DIM_C * NTOK)

// ---------------- vector types ----------------
typedef _Float16 h8  __attribute__((ext_vector_type(8)));
typedef _Float16 h16 __attribute__((ext_vector_type(16)));
typedef float    f8  __attribute__((ext_vector_type(8)));
typedef int      v4i __attribute__((ext_vector_type(4)));

union Frag { h16 v; h8 h[2]; };

static __device__ __forceinline__ f8 wmma_f16(const Frag& a, const Frag& b, f8 c) {
  return __builtin_amdgcn_wmma_f32_16x16x32_f16(false, a.v, false, b.v,
                                                (short)0, c, false, false);
}

// gfx1250 async global->LDS path (ASYNCcnt-tracked), guarded: probe via __has_builtin.
// Builtin signature (from hipcc diagnostic): (v4i AS1*, v4i AS3*, imm offset, imm cpol)
#if __has_builtin(__builtin_amdgcn_global_load_async_to_lds_b128) && \
    __has_builtin(__builtin_amdgcn_s_wait_asynccnt)
#define GEMM_ASYNC 1
typedef __attribute__((address_space(1))) v4i gv4i;
typedef __attribute__((address_space(3))) v4i lv4i;
static __device__ __forceinline__ void async_lds_b128(const void* g, void* l) {
  __builtin_amdgcn_global_load_async_to_lds_b128((gv4i*)g, (lv4i*)l, 0, 0);
}
#else
#define GEMM_ASYNC 0
#endif

// ---------------- f32 -> f16 convert ----------------
__global__ void cvt_f16_kernel(const float* __restrict__ a, _Float16* __restrict__ b, int n) {
  int i = blockIdx.x * blockDim.x + threadIdx.x;
  if (i < n) b[i] = (_Float16)a[i];
}

// ---------------- depthwise conv3d 3x3x3 + bias + residual -> tokens ----------------
__global__ void conv_pe_kernel(const float* __restrict__ x, const float* __restrict__ pw,
                               const float* __restrict__ pb, float* __restrict__ t) {
  int idx = blockIdx.x * blockDim.x + threadIdx.x;
  if (idx >= TOTAL) return;
  int w = idx % 14;
  int h = (idx / 14) % 14;
  int d = (idx / 196) % 8;
  int c = (idx / NTOK) % DIM_C;
  int b = idx / (NTOK * DIM_C);
  const float* xb = x + ((size_t)b * DIM_C + c) * (size_t)NTOK; // x[b,c,:,:,:]
  const float* wc = pw + c * 27;
  float s = pb[c];
  #pragma unroll
  for (int kd = 0; kd < 3; ++kd) {
    int zd = d + kd - 1;
    if (zd < 0 || zd >= 8) continue;
    #pragma unroll
    for (int kh = 0; kh < 3; ++kh) {
      int zh = h + kh - 1;
      if (zh < 0 || zh >= 14) continue;
      #pragma unroll
      for (int kw = 0; kw < 3; ++kw) {
        int zw = w + kw - 1;
        if (zw < 0 || zw >= 14) continue;
        s += xb[(zd * 14 + zh) * 14 + zw] * wc[kd * 9 + kh * 3 + kw];
      }
    }
  }
  int n = (d * 14 + h) * 14 + w;
  t[((size_t)b * NTOK + n) * DIM_C + c] = xb[(d * 14 + h) * 14 + w] + s;
}

// ---------------- LayerNorm (block per token) -> f16 ----------------
__global__ __launch_bounds__(256)
void ln_kernel(const float* __restrict__ t, const float* __restrict__ wln,
               const float* __restrict__ bln, _Float16* __restrict__ xh) {
  __shared__ float red[256];
  const int tok = blockIdx.x;
  const int tid = threadIdx.x;
  const float* row = t + (size_t)tok * DIM_C;
  float v[3]; float s = 0.0f;
  #pragma unroll
  for (int i = 0; i < 3; ++i) { v[i] = row[tid + 256 * i]; s += v[i]; }
  red[tid] = s; __syncthreads();
  for (int o = 128; o > 0; o >>= 1) { if (tid < o) red[tid] += red[tid + o]; __syncthreads(); }
  float mu = red[0] * (1.0f / DIM_C);
  __syncthreads();
  float sq = 0.0f;
  #pragma unroll
  for (int i = 0; i < 3; ++i) { float dlt = v[i] - mu; sq += dlt * dlt; }
  red[tid] = sq; __syncthreads();
  for (int o = 128; o > 0; o >>= 1) { if (tid < o) red[tid] += red[tid + o]; __syncthreads(); }
  float rstd = rsqrtf(red[0] * (1.0f / DIM_C) + 1e-5f);
  #pragma unroll
  for (int i = 0; i < 3; ++i) {
    int c = tid + 256 * i;
    xh[(size_t)tok * DIM_C + c] = (_Float16)((v[i] - mu) * rstd * wln[c] + bln[c]);
  }
}

// ---------------- WMMA GEMM:  out = X[M,K] * W[N,K]^T  (torch Linear layout) ----------------
// block = 256 thr = 8 waves; block tile 32(M) x 256(N); wave tile 16 x 64.
// A staged in LDS, K-depth 64, double-buffered via async global->LDS (one barrier / iter).
// EPI: 0 = store f16 (no bias) | 1 = +bias, residual-add into f32 tres | 2 = +bias, GELU -> f16
template <int EPI>
__global__ __launch_bounds__(256)
void gemm_wmma_kernel(const _Float16* __restrict__ X, const _Float16* __restrict__ W,
                      const float* __restrict__ bias, _Float16* __restrict__ outh,
                      float* __restrict__ tres, int K, int Nout) {
  __shared__ _Float16 Atile[2][32][64];   // 8 KB, double-buffered
  const int tid  = threadIdx.x;
  const int lane = tid & 31;
  const int wv   = tid >> 5;
  const int mi   = wv & 1;        // 0..1 : M sub-tile
  const int ni   = wv >> 1;       // 0..3 : N sub-tile (64 wide)
  const int mblk = blockIdx.x * 32;
  const int nblk = blockIdx.y * 256;
  const int m15  = lane & 15;
  const int kbA  = (lane >> 4) * 8;   // A-fragment K base
  const int kbB  = (lane >> 4) * 8;   // B-fragment K base
  const int mrowA = mi * 16 + m15;
  // cooperative stage geometry: 256 thr x 16B = 4KB = 32 rows x 64 halfs
  const int srow = tid >> 3;
  const int scol = (tid & 7) * 8;

  const int KT = K >> 6;  // K / 64
  f8 acc[4] = {};

  // prologue: stage tile 0 into buffer 0
  {
    const _Float16* src = &X[(size_t)(mblk + srow) * K + scol];
    _Float16* dst = &Atile[0][srow][scol];
#if GEMM_ASYNC
    async_lds_b128(src, dst);
    __builtin_amdgcn_s_wait_asynccnt(0);
#else
    *(h8*)dst = *(const h8*)src;
#endif
  }
  __syncthreads();

  for (int kt = 0; kt < KT; ++kt) {
    const int buf = kt & 1;
#if GEMM_ASYNC
    if (kt + 1 < KT) {  // overlap: async-copy next tile while computing this one
      const _Float16* src = &X[(size_t)(mblk + srow) * K + (kt + 1) * 64 + scol];
      async_lds_b128(src, &Atile[buf ^ 1][srow][scol]);
    }
#endif
    // A fragments for the two 32-deep K sub-steps (ds_load_b128 x4)
    Frag a0, a1;
    a0.h[0] = *(const h8*)&Atile[buf][mrowA][kbA];
    a0.h[1] = *(const h8*)&Atile[buf][mrowA][kbA + 16];
    a1.h[0] = *(const h8*)&Atile[buf][mrowA][32 + kbA];
    a1.h[1] = *(const h8*)&Atile[buf][mrowA][32 + kbA + 16];
    #pragma unroll
    for (int tl = 0; tl < 4; ++tl) {
      int n = nblk + ni * 64 + tl * 16 + m15;
      const _Float16* wp = &W[(size_t)n * K + kt * 64 + kbB];
      __builtin_prefetch(wp + 64, 0, 1);  // next K tile of this weight row
      Frag b0, b1;
      b0.h[0] = *(const h8*)wp;
      b0.h[1] = *(const h8*)(wp + 16);
      b1.h[0] = *(const h8*)(wp + 32);
      b1.h[1] = *(const h8*)(wp + 48);
      acc[tl] = wmma_f16(a0, b0, acc[tl]);
      acc[tl] = wmma_f16(a1, b1, acc[tl]);
    }
#if GEMM_ASYNC
    __builtin_amdgcn_s_wait_asynccnt(0);  // my async stores for next tile landed in LDS
    __syncthreads();                      // everyone done reading cur + staged next
#else
    __syncthreads();                      // all readers done before overwrite
    if (kt + 1 < KT) {
      const _Float16* src = &X[(size_t)(mblk + srow) * K + (kt + 1) * 64 + scol];
      *(h8*)&Atile[buf ^ 1][srow][scol] = *(const h8*)src;
    }
    __syncthreads();
#endif
  }

  #pragma unroll
  for (int tl = 0; tl < 4; ++tl) {
    int col = nblk + ni * 64 + tl * 16 + m15;
    #pragma unroll
    for (int r = 0; r < 8; ++r) {
      int row = mblk + mi * 16 + ((lane >> 4) << 3) + r;
      float vacc = acc[tl][r];
      if (EPI == 0) {
        outh[(size_t)row * Nout + col] = (_Float16)vacc;
      } else if (EPI == 1) {
        tres[(size_t)row * Nout + col] += vacc + bias[col];
      } else {
        float z = vacc + bias[col];
        float g = 0.5f * z * (1.0f + erff(z * 0.70710678118f));
        outh[(size_t)row * Nout + col] = (_Float16)g;
      }
    }
  }
}

// ---------------- flash attention (wave per 16-query block) ----------------
// qkv rows: [q(12*64) | k(12*64) | v(12*64)]  f16. 49 key blocks of 32.
__global__ __launch_bounds__(256)
void attn_kernel(const _Float16* __restrict__ qkv, _Float16* __restrict__ o) {
  __shared__ _Float16 Plds[8][16][32];   // per-wave P tile (16 q x 32 keys)
  __shared__ _Float16 Vt[8][64][32];     // per-wave transposed V block (d x key)
  const int tid  = threadIdx.x;
  const int lane = tid & 31;
  const int wv   = tid >> 5;
  const int m15  = lane & 15;
  const int kbA  = (lane >> 4) * 8;
  const int kbB  = (lane >> 4) * 8;

  int bh   = blockIdx.x / 13;          // (b, head)
  int qgrp = blockIdx.x % 13;
  int b    = bh / NHEAD;
  int head = bh % NHEAD;
  int qb   = qgrp * 8 + wv; if (qb > 97) qb = 97;   // clamp: duplicate work, no divergence
  const int q0 = qb * 16;

  const _Float16* base = qkv + (size_t)b * NTOK * H3 + head * DHEAD;

  Frag qf[2];
  {
    const _Float16* qp = base + (size_t)(q0 + m15) * H3;
    qf[0].h[0] = *(const h8*)(qp + kbA);
    qf[0].h[1] = *(const h8*)(qp + kbA + 16);
    qf[1].h[0] = *(const h8*)(qp + 32 + kbA);
    qf[1].h[1] = *(const h8*)(qp + 32 + kbA + 16);
  }

  float rmax[8], rsum[8];
  #pragma unroll
  for (int r = 0; r < 8; ++r) { rmax[r] = -1e30f; rsum[r] = 0.0f; }
  f8 acc[4] = {};

  for (int mb = 0; mb < 49; ++mb) {
    const int key0 = mb * 32;
    // ---- S = Q * K^T (two 16x16 tiles over 32 keys) ----
    f8 s[2];
    #pragma unroll
    for (int hf = 0; hf < 2; ++hf) {
      const _Float16* kp = base + (size_t)(key0 + hf * 16 + m15) * H3 + DIM_C;
      f8 sa = {};
      #pragma unroll
      for (int dc = 0; dc < 2; ++dc) {
        Frag kb_;
        kb_.h[0] = *(const h8*)(kp + dc * 32 + kbB);
        kb_.h[1] = *(const h8*)(kp + dc * 32 + kbB + 16);
        sa = wmma_f16(qf[dc], kb_, sa);
      }
      s[hf] = sa;
    }
    // ---- online softmax (row reductions across 16-lane halves) ----
    float p0[8], p1[8];
    #pragma unroll
    for (int r = 0; r < 8; ++r) {
      float s0 = s[0][r] * 0.125f;   // scale = 1/sqrt(64)
      float s1 = s[1][r] * 0.125f;
      float mr = fmaxf(s0, s1);
      #pragma unroll
      for (int off = 1; off < 16; off <<= 1) mr = fmaxf(mr, __shfl_xor(mr, off, 16));
      float nm   = fmaxf(rmax[r], mr);
      float corr = __expf(rmax[r] - nm);
      float e0 = __expf(s0 - nm), e1 = __expf(s1 - nm);
      float rs = e0 + e1;
      #pragma unroll
      for (int off = 1; off < 16; off <<= 1) rs += __shfl_xor(rs, off, 16);
      rsum[r] = rsum[r] * corr + rs;
      rmax[r] = nm;
      #pragma unroll
      for (int tl = 0; tl < 4; ++tl) acc[tl][r] *= corr;
      p0[r] = e0; p1[r] = e1;
    }
    __syncthreads();
    // ---- stage P (D-layout -> A-layout via LDS) ----
    #pragma unroll
    for (int r = 0; r < 8; ++r) {
      int row = ((lane >> 4) << 3) + r;
      Plds[wv][row][m15]      = (_Float16)p0[r];
      Plds[wv][row][16 + m15] = (_Float16)p1[r];
    }
    // ---- stage V transposed: lane owns one key row ----
    {
      const _Float16* vp = base + (size_t)(key0 + lane) * H3 + 2 * DIM_C;
      #pragma unroll
      for (int d8 = 0; d8 < 8; ++d8) {
        h8 vv = *(const h8*)(vp + d8 * 8);
        #pragma unroll
        for (int j = 0; j < 8; ++j) Vt[wv][d8 * 8 + j][lane] = vv[j];
      }
    }
    __syncthreads();
    // ---- O += P * V ----
    Frag pf;
    pf.h[0] = *(const h8*)&Plds[wv][m15][kbA];
    pf.h[1] = *(const h8*)&Plds[wv][m15][kbA + 16];
    #pragma unroll
    for (int tl = 0; tl < 4; ++tl) {
      Frag vf;
      vf.h[0] = *(const h8*)&Vt[wv][tl * 16 + m15][kbB];
      vf.h[1] = *(const h8*)&Vt[wv][tl * 16 + m15][kbB + 16];
      acc[tl] = wmma_f16(pf, vf, acc[tl]);
    }
  }
  // ---- normalize and write O (f16, [B,N,C] layout) ----
  #pragma unroll
  for (int tl = 0; tl < 4; ++tl) {
    #pragma unroll
    for (int r = 0; r < 8; ++r) {
      int row = q0 + ((lane >> 4) << 3) + r;
      float vo = acc[tl][r] / rsum[r];
      o[((size_t)b * NTOK + row) * DIM_C + head * DHEAD + tl * 16 + m15] = (_Float16)vo;
    }
  }
}

// ---------------- tokens [B,N,C] f32 -> output [B,C,D,H,W] ----------------
__global__ void out_kernel(const float* __restrict__ t, float* __restrict__ out) {
  int idx = blockIdx.x * blockDim.x + threadIdx.x;
  if (idx >= TOTAL) return;
  int n = idx % NTOK;
  int c = (idx / NTOK) % DIM_C;
  int b = idx / (NTOK * DIM_C);
  out[idx] = t[((size_t)b * NTOK + n) * DIM_C + c];
}

// ---------------- workspace layout ----------------
static constexpr size_t algn(size_t x) { return (x + 255) & ~size_t(255); }
static constexpr size_t T_OFF   = 0;
static constexpr size_t XH_OFF  = T_OFF   + algn((size_t)MTOK * DIM_C * 4);
static constexpr size_t QKV_OFF = XH_OFF  + algn((size_t)MTOK * DIM_C * 2);
static constexpr size_t O_OFF   = QKV_OFF + algn((size_t)MTOK * H3 * 2);
static constexpr size_t HH_OFF  = O_OFF   + algn((size_t)MTOK * DIM_C * 2);
static constexpr size_t WQ_OFF  = HH_OFF  + algn((size_t)MTOK * HID * 2);
static constexpr size_t WP_OFF  = WQ_OFF  + algn((size_t)H3 * DIM_C * 2);
static constexpr size_t W1_OFF  = WP_OFF  + algn((size_t)DIM_C * DIM_C * 2);
static constexpr size_t W2_OFF  = W1_OFF  + algn((size_t)HID * DIM_C * 2);

extern "C" void kernel_launch(void* const* d_in, const int* in_sizes, int n_in,
                              void* d_out, int out_size, void* d_ws, size_t ws_size,
                              hipStream_t stream) {
  const float* x      = (const float*)d_in[0];
  const float* pos_w  = (const float*)d_in[1];
  const float* pos_b  = (const float*)d_in[2];
  const float* ln1_w  = (const float*)d_in[3];
  const float* ln1_b  = (const float*)d_in[4];
  const float* qkv_w  = (const float*)d_in[5];
  const float* proj_w = (const float*)d_in[6];
  const float* proj_b = (const float*)d_in[7];
  const float* ln2_w  = (const float*)d_in[8];
  const float* ln2_b  = (const float*)d_in[9];
  const float* fc1_w  = (const float*)d_in[10];
  const float* fc1_b  = (const float*)d_in[11];
  const float* fc2_w  = (const float*)d_in[12];
  const float* fc2_b  = (const float*)d_in[13];
  float* out = (float*)d_out;
  char*  ws  = (char*)d_ws;

  float*    t    = (float*)   (ws + T_OFF);
  _Float16* xh   = (_Float16*)(ws + XH_OFF);
  _Float16* qkvh = (_Float16*)(ws + QKV_OFF);
  _Float16* oh   = (_Float16*)(ws + O_OFF);
  _Float16* hh   = (_Float16*)(ws + HH_OFF);
  _Float16* wq   = (_Float16*)(ws + WQ_OFF);
  _Float16* wp   = (_Float16*)(ws + WP_OFF);
  _Float16* w1   = (_Float16*)(ws + W1_OFF);
  _Float16* w2   = (_Float16*)(ws + W2_OFF);

  // 1) weights -> f16 (stay resident in 192MB L2)
  cvt_f16_kernel<<<(H3 * DIM_C + 255) / 256, 256, 0, stream>>>(qkv_w, wq, H3 * DIM_C);
  cvt_f16_kernel<<<(DIM_C * DIM_C + 255) / 256, 256, 0, stream>>>(proj_w, wp, DIM_C * DIM_C);
  cvt_f16_kernel<<<(HID * DIM_C + 255) / 256, 256, 0, stream>>>(fc1_w, w1, HID * DIM_C);
  cvt_f16_kernel<<<(DIM_C * HID + 255) / 256, 256, 0, stream>>>(fc2_w, w2, DIM_C * HID);

  // 2) depthwise conv PE + bias + residual, into token-major f32
  conv_pe_kernel<<<(TOTAL + 255) / 256, 256, 0, stream>>>(x, pos_w, pos_b, t);

  // 3) LN1 -> f16
  ln_kernel<<<MTOK, 256, 0, stream>>>(t, ln1_w, ln1_b, xh);

  // 4) QKV gemm (6272x768 @ 768x2304)
  gemm_wmma_kernel<0><<<dim3(MTOK / 32, H3 / 256), 256, 0, stream>>>(
      xh, wq, nullptr, qkvh, nullptr, DIM_C, H3);

  // 5) flash attention
  attn_kernel<<<BATCH * NHEAD * 13, 256, 0, stream>>>(qkvh, oh);

  // 6) proj + bias + residual into t
  gemm_wmma_kernel<1><<<dim3(MTOK / 32, DIM_C / 256), 256, 0, stream>>>(
      oh, wp, proj_b, nullptr, t, DIM_C, DIM_C);

  // 7) LN2 -> f16
  ln_kernel<<<MTOK, 256, 0, stream>>>(t, ln2_w, ln2_b, xh);

  // 8) FC1 + bias + GELU -> f16
  gemm_wmma_kernel<2><<<dim3(MTOK / 32, HID / 256), 256, 0, stream>>>(
      xh, w1, fc1_b, hh, nullptr, DIM_C, HID);

  // 9) FC2 + bias + residual into t
  gemm_wmma_kernel<1><<<dim3(MTOK / 32, DIM_C / 256), 256, 0, stream>>>(
      hh, w2, fc2_b, nullptr, t, HID, DIM_C);

  // 10) token-major -> NCDHW
  out_kernel<<<(TOTAL + 255) / 256, 256, 0, stream>>>(t, out);
}
